// Model_70119636074637
// MI455X (gfx1250) — matrix operand order repaired and checked by reading, not compiled, not run on hardware
//
#include <hip/hip_runtime.h>
#include <hip/hip_bf16.h>
#include <math.h>

// ---------------------------------------------------------------------------
// Types / WMMA helpers (CDNA5 gfx1250, wave32, v_wmma_f32_16x16x32_bf16)
// ---------------------------------------------------------------------------
typedef __bf16 bf16;
typedef __bf16 v16bf __attribute__((ext_vector_type(16)));
typedef __bf16 v8bf  __attribute__((ext_vector_type(8)));
typedef float  v8f   __attribute__((ext_vector_type(8)));

#define DEV static __device__ __forceinline__

DEV v8f wmma_bf(v16bf a, v16bf b, v8f c) {
  // 8 args: (neg_a, A, neg_b, B, c_mod, C, reuse_a, reuse_b)
  return __builtin_amdgcn_wmma_f32_16x16x32_bf16(false, a, false, b, (short)0, c,
                                                 false, false);
}

DEV v8f v8f_zero() {
  v8f c = {0.f, 0.f, 0.f, 0.f, 0.f, 0.f, 0.f, 0.f};
  return c;
}

DEV v8bf v8bf_zero() {
  union { v8bf v; int i[4]; } z;
  z.i[0] = z.i[1] = z.i[2] = z.i[3] = 0;
  return z.v;
}

// A-fragment (16x32 bf16, MxK) from row-major [ld] matrix, tile origin `base`.
// Layout (ISA 7.12.2): lane L: m=L&15, half=L>>4.
//   regs 0..7  <-> K = half*8 + (0..7)
//   regs 8..15 <-> K = 16 + half*8 + (0..7)
// -> two contiguous 8-element (16B) loads per lane.
DEV v16bf load_a_frag(const bf16* base, int ld, int lane) {
  int m = lane & 15, half = lane >> 4;
  const bf16* p = base + (size_t)m * ld;
  union { v16bf v; v8bf h[2]; } u;
  u.h[0] = *(const v8bf*)(p + half * 8);
  u.h[1] = *(const v8bf*)(p + 16 + half * 8);
  return u.v;
}

// B-fragment (32x16 bf16, KxN) where B[k][n] = W[n0+n][k0+k] (W row-major, ld).
// Layout: lane L: n=L&15, half=L>>4; regs 0..7 <-> K = half*8 + (0..7),
// regs 8..15 <-> K = 16 + half*8 + (0..7) (same as A).
DEV v16bf load_b_frag(const bf16* base, int ld, int lane) {
  int n = lane & 15, half = lane >> 4;
  const bf16* p = base + (size_t)n * ld;
  union { v16bf v; v8bf h[2]; } u;
  u.h[0] = *(const v8bf*)(p + half * 8);
  u.h[1] = *(const v8bf*)(p + 16 + half * 8);
  return u.v;
}

// Problem dimensions
#define B_  32
#define T_  24
#define U_  300
#define H_  128
#define E_  168
#define CG_ 32
#define BU  (B_ * U_)          // 9600
#define MT_ (B_ * T_ * U_)     // 230400 rows of (b,t,u)

// ---------------------------------------------------------------------------
// Weight conversion kernels (f32 -> bf16, optionally zero-padded)
// ---------------------------------------------------------------------------
__global__ void cvt_kernel(const float* __restrict__ src, bf16* __restrict__ dst, int n) {
  int i = blockIdx.x * 256 + threadIdx.x;
  if (i < n) dst[i] = (bf16)src[i];
}

__global__ void cvt_pad_kernel(const float* __restrict__ src, bf16* __restrict__ dst,
                               int rows, int cols, int dcols) {
  int i = blockIdx.x * 256 + threadIdx.x;
  if (i >= rows * dcols) return;
  int r = i / dcols, c = i % dcols;
  dst[i] = (bf16)((c < cols) ? src[r * cols + c] : 0.f);
}

// fc1_w [296][328] -> padded bf16 [304][352]: cols 0..295 = Xn part,
// 296..319 zero (matches Xn zero pad), 320..351 = g part (real cols 296..327).
__global__ void cvt_fc1_kernel(const float* __restrict__ w, bf16* __restrict__ dst) {
  int i = blockIdx.x * 256 + threadIdx.x;
  if (i >= 304 * 352) return;
  int r = i / 352, c = i % 352;
  float v = 0.f;
  if (r < 296) {
    if (c < 296) v = w[r * 328 + c];
    else if (c >= 320) v = w[r * 328 + 296 + (c - 320)];
  }
  dst[i] = (bf16)v;
}

__global__ void zero_f32_kernel(float* __restrict__ p, int n) {
  int i = blockIdx.x * 256 + threadIdx.x;
  if (i < n) p[i] = 0.f;
}

// ---------------------------------------------------------------------------
// K1: seq1 — feats[b,t,u,128] = relu(W2 @ relu(x*w1+b1) + b2), WMMA K=64
// one wave per 16-row M tile; h1 staged in LDS as bf16 A tiles
// ---------------------------------------------------------------------------
__global__ __launch_bounds__(256) void seq1_kernel(
    const float* __restrict__ x, const float* __restrict__ w1,
    const float* __restrict__ b1, const bf16* __restrict__ w2bf,
    const float* __restrict__ b2, bf16* __restrict__ feats) {
  __shared__ __align__(16) bf16 h1[8][16 * 64];
  int w = threadIdx.x >> 5, lane = threadIdx.x & 31;
  int tile = blockIdx.x * 8 + w;        // 14400 tiles total
  int i0 = tile * 16;
  for (int idx = lane; idx < 16 * 64; idx += 32) {
    int row = idx >> 6, c = idx & 63;
    float xv = x[i0 + row];
    h1[w][row * 64 + c] = (bf16)fmaxf(xv * w1[c] + b1[c], 0.f);
  }
  __syncthreads();
  int n = lane & 15, half = lane >> 4;
  v16bf a0 = load_a_frag(&h1[w][0], 64, lane);
  v16bf a1 = load_a_frag(&h1[w][32], 64, lane);
  #pragma unroll
  for (int nt = 0; nt < 8; ++nt) {
    v8f acc = v8f_zero();
    acc = wmma_bf(a0, load_b_frag(w2bf + (nt * 16) * 64, 64, lane), acc);
    acc = wmma_bf(a1, load_b_frag(w2bf + (nt * 16) * 64 + 32, 64, lane), acc);
    int c = nt * 16 + n;
    float bias = b2[c];
    #pragma unroll
    for (int r = 0; r < 8; ++r) {
      float v = fmaxf(acc[r] + bias, 0.f);
      feats[(size_t)(i0 + half * 8 + r) * 128 + c] = (bf16)v;
    }
  }
}

// ---------------------------------------------------------------------------
// K2: fused per-node GRU. Block = node u (300 blocks, 8 waves).
// Wave w owns h columns [16w, 16w+16). Per t: 48 WMMAs/wave, gate math on
// C-fragments (lane = column -> gate columns align across r/z/n accumulators),
// h shared through LDS between steps. outs (=h after each step) -> global bf16.
// ---------------------------------------------------------------------------
__global__ __launch_bounds__(256) void gru_kernel(
    const bf16* __restrict__ feats, const bf16* __restrict__ wih,
    const bf16* __restrict__ whh, const float* __restrict__ bih,
    const float* __restrict__ bhh, bf16* __restrict__ outs) {
  int u = blockIdx.x;
  int lane = threadIdx.x & 31;
  int w = threadIdx.x >> 5;
  __shared__ __align__(16) bf16 hbuf[32 * 128];
  for (int i = threadIdx.x; i < 32 * 128; i += 256) hbuf[i] = (bf16)0.f;
  __syncthreads();

  float hreg[2][8];
  #pragma unroll
  for (int mt = 0; mt < 2; ++mt)
    #pragma unroll
    for (int r = 0; r < 8; ++r) hreg[mt][r] = 0.f;

  int n = lane & 15, half = lane >> 4;
  int col = w * 16 + n;                    // output H-column this lane carries
  float bi[3], bh[3];
  #pragma unroll
  for (int g = 0; g < 3; ++g) {
    bi[g] = bih[u * 384 + g * 128 + col];
    bh[g] = bhh[u * 384 + g * 128 + col];
  }
  const bf16* wihu = wih + (size_t)u * 384 * 128;
  const bf16* whhu = whh + (size_t)u * 384 * 128;

  for (int t = 0; t < T_; ++t) {
    v16bf ah[2][4];
    #pragma unroll
    for (int mt = 0; mt < 2; ++mt)
      #pragma unroll
      for (int ks = 0; ks < 4; ++ks)
        ah[mt][ks] = load_a_frag(hbuf + (mt * 16) * 128 + ks * 32, 128, lane);
    __syncthreads();   // all waves done reading h before anyone writes h_new

    #pragma unroll
    for (int mt = 0; mt < 2; ++mt) {
      v8f acc[6];
      #pragma unroll
      for (int g = 0; g < 6; ++g) acc[g] = v8f_zero();
      #pragma unroll
      for (int ks = 0; ks < 4; ++ks) {
        const bf16* fa =
            feats + (((size_t)(mt * 16) * T_ + t) * U_ + u) * 128 + ks * 32;
        v16bf af = load_a_frag(fa, T_ * U_ * 128, lane);  // row stride = batch
        v16bf ahk = ah[mt][ks];
        #pragma unroll
        for (int g = 0; g < 3; ++g) {
          v16bf bI = load_b_frag(wihu + (g * 128 + w * 16) * 128 + ks * 32, 128, lane);
          acc[g] = wmma_bf(af, bI, acc[g]);
          v16bf bH = load_b_frag(whhu + (g * 128 + w * 16) * 128 + ks * 32, 128, lane);
          acc[3 + g] = wmma_bf(ahk, bH, acc[3 + g]);
        }
      }
      #pragma unroll
      for (int r = 0; r < 8; ++r) {
        float xr = acc[0][r] + bi[0], xz = acc[1][r] + bi[1], xn = acc[2][r] + bi[2];
        float hr = acc[3][r] + bh[0], hz = acc[4][r] + bh[1], hn = acc[5][r] + bh[2];
        float rg = 1.f / (1.f + __expf(-(xr + hr)));
        float zg = 1.f / (1.f + __expf(-(xz + hz)));
        float ng = tanhf(xn + rg * hn);
        float hnew = (1.f - zg) * ng + zg * hreg[mt][r];
        hreg[mt][r] = hnew;
        int b = mt * 16 + half * 8 + r;
        bf16 hb = (bf16)hnew;
        hbuf[b * 128 + col] = hb;
        outs[(((size_t)t * B_ + b) * U_ + u) * 128 + col] = hb;
      }
    }
    __syncthreads();   // h_new visible before next step's reads
  }
}

// ---------------------------------------------------------------------------
// K3a: temporal attention scores: tanh([hid;outs_t] @ attn_w^T + b) . v
// M = T*B*U flat rows, K = 256 (hid||out), fused row-dot with attn_v via
// cross-lane reduction within each 16-lane half.
// ---------------------------------------------------------------------------
__global__ __launch_bounds__(256) void score_kernel(
    const bf16* __restrict__ outs, const bf16* __restrict__ attnw_bf,
    const float* __restrict__ attn_b, const float* __restrict__ attn_v,
    float* __restrict__ scores) {
  int wave = (blockIdx.x * 256 + threadIdx.x) >> 5;   // 14400 waves
  int lane = threadIdx.x & 31;
  int i0 = wave * 16;
  int hid0 = (T_ - 1) * BU + (i0 % BU);               // hid == outs[T-1]
  v16bf af[8];
  #pragma unroll
  for (int ks = 0; ks < 4; ++ks)
    af[ks] = load_a_frag(outs + (size_t)hid0 * 128 + ks * 32, 128, lane);
  #pragma unroll
  for (int ks = 0; ks < 4; ++ks)
    af[4 + ks] = load_a_frag(outs + (size_t)i0 * 128 + ks * 32, 128, lane);
  int n = lane & 15, half = lane >> 4;
  float rs[8];
  #pragma unroll
  for (int r = 0; r < 8; ++r) rs[r] = 0.f;
  #pragma unroll
  for (int nt = 0; nt < 8; ++nt) {
    v8f acc = v8f_zero();
    #pragma unroll
    for (int ks = 0; ks < 8; ++ks)
      acc = wmma_bf(af[ks], load_b_frag(attnw_bf + (nt * 16) * 256 + ks * 32, 256, lane), acc);
    int c = nt * 16 + n;
    float bb = attn_b[c], vv = attn_v[c];
    #pragma unroll
    for (int r = 0; r < 8; ++r) rs[r] += tanhf(acc[r] + bb) * vv;
  }
  #pragma unroll
  for (int r = 0; r < 8; ++r) {
    float v = rs[r];
    for (int off = 1; off < 16; off <<= 1) v += __shfl_xor(v, off, 32);
    if (n == 0) scores[i0 + half * 8 + r] = v;
  }
}

// ---------------------------------------------------------------------------
// K3b: softmax over T + weighted sum of outs + residual + layernorm
// ---------------------------------------------------------------------------
__global__ __launch_bounds__(128) void timeattn_kernel(
    const float* __restrict__ scores, const bf16* __restrict__ outs,
    const float* __restrict__ ln_w, const float* __restrict__ ln_b,
    bf16* __restrict__ wres) {
  int bu = blockIdx.x;
  int c = threadIdx.x;
  __shared__ float wt[T_];
  __shared__ float red[128];
  if (c < T_) wt[c] = scores[c * BU + bu];
  __syncthreads();
  if (c == 0) {
    float mx = -1e30f;
    for (int t = 0; t < T_; ++t) mx = fmaxf(mx, wt[t]);
    float s = 0.f;
    for (int t = 0; t < T_; ++t) { wt[t] = __expf(wt[t] - mx); s += wt[t]; }
    float inv = 1.f / s;
    for (int t = 0; t < T_; ++t) wt[t] *= inv;
  }
  __syncthreads();
  float acc = 0.f;
  for (int t = 0; t < T_; ++t)
    acc += wt[t] * (float)outs[((size_t)t * BU + bu) * 128 + c];
  float hid = (float)outs[((size_t)(T_ - 1) * BU + bu) * 128 + c];
  float v = acc + hid;
  red[c] = v; __syncthreads();
  for (int s = 64; s > 0; s >>= 1) { if (c < s) red[c] += red[c + s]; __syncthreads(); }
  float mu = red[0] / 128.f;
  __syncthreads();
  float d = v - mu;
  red[c] = d * d; __syncthreads();
  for (int s = 64; s > 0; s >>= 1) { if (c < s) red[c] += red[c + s]; __syncthreads(); }
  float var = red[0] / 128.f;
  wres[(size_t)bu * 128 + c] =
      (bf16)(d * rsqrtf(var + 1e-5f) * ln_w[c] + ln_b[c]);
}

// ---------------------------------------------------------------------------
// K4a: q/k projections (M=9600, K=128, N=128), bias added, bf16 out
// ---------------------------------------------------------------------------
__global__ __launch_bounds__(256) void qk_kernel(
    const bf16* __restrict__ wres, const bf16* __restrict__ mhaw_bf,
    const float* __restrict__ mha_b, bf16* __restrict__ qb, bf16* __restrict__ kb) {
  int wave = (blockIdx.x * 256 + threadIdx.x) >> 5;   // 1200 waves
  int lane = threadIdx.x & 31;
  int which = wave / 600;                             // 0 = q, 1 = k
  int i0 = (wave % 600) * 16;
  v16bf af[4];
  #pragma unroll
  for (int ks = 0; ks < 4; ++ks)
    af[ks] = load_a_frag(wres + (size_t)i0 * 128 + ks * 32, 128, lane);
  const bf16* W = mhaw_bf + (size_t)which * 128 * 128;
  const float* bias = mha_b + which * 128;
  bf16* dst = which ? kb : qb;
  int n = lane & 15, half = lane >> 4;
  #pragma unroll
  for (int nt = 0; nt < 8; ++nt) {
    v8f acc = v8f_zero();
    #pragma unroll
    for (int ks = 0; ks < 4; ++ks)
      acc = wmma_bf(af[ks], load_b_frag(W + (nt * 16) * 128 + ks * 32, 128, lane), acc);
    int c = nt * 16 + n;
    float bv = bias[c];
    #pragma unroll
    for (int r = 0; r < 8; ++r)
      dst[(size_t)(i0 + half * 8 + r) * 128 + c] = (bf16)(acc[r] + bv);
  }
}

// ---------------------------------------------------------------------------
// K4b: MHA logits (K=16 padded to 32) + row softmax over 300 cols + atomic
// mean into attention[U,U]. One wave per (b, head, 16-row tile).
// ---------------------------------------------------------------------------
__global__ __launch_bounds__(128) void mha_attn_kernel(
    const bf16* __restrict__ qb, const bf16* __restrict__ kb,
    float* __restrict__ attention) {
  int gw = (blockIdx.x * 128 + threadIdx.x) >> 5;   // 32*8*19 = 4864 waves
  int lane = threadIdx.x & 31;
  int mt = gw % 19;
  int h = (gw / 19) % 8;
  int b = gw / (19 * 8);
  int u0 = mt * 16;
  int n = lane & 15, half = lane >> 4;

  v16bf aq;
  {
    int row = u0 + (lane & 15); if (row > U_ - 1) row = U_ - 1;
    const bf16* p = qb + ((size_t)b * U_ + row) * 128 + h * 16;
    union { v16bf v; v8bf hh[2]; } Uv;
    Uv.hh[0] = *(const v8bf*)(p + half * 8);   // K = half*8..+7 (<16, real)
    Uv.hh[1] = v8bf_zero();                    // K >= 16 -> zero pad
    aq = Uv.v;
  }
  v8f l[19];
  #pragma unroll
  for (int nt = 0; nt < 19; ++nt) {
    union { v16bf v; v8bf hh[2]; } Uv;
    int colv = nt * 16 + n; if (colv > U_ - 1) colv = U_ - 1;
    {                                          // K = half*8..+7 (<16, real); K >= 16 -> 0
      const bf16* p = kb + ((size_t)b * U_ + colv) * 128 + h * 16;
      Uv.hh[0] = *(const v8bf*)(p + half * 8);
      Uv.hh[1] = v8bf_zero();
    }
    l[nt] = wmma_bf(aq, Uv.v, v8f_zero());
  }
  const float scale = 0.25f;                   // 1/sqrt(HD=16)
  const float rowscale = 1.f / (32.f * 8.f);   // mean over batch and heads
  #pragma unroll
  for (int r = 0; r < 8; ++r) {
    int u = u0 + half * 8 + r;
    float mx = -3.0e38f;
    #pragma unroll
    for (int nt = 0; nt < 19; ++nt) {
      int v = nt * 16 + n;
      float x = (v < U_) ? l[nt][r] * scale : -3.0e38f;
      l[nt][r] = x;
      mx = fmaxf(mx, x);
    }
    for (int off = 1; off < 16; off <<= 1) mx = fmaxf(mx, __shfl_xor(mx, off, 32));
    float s = 0.f;
    #pragma unroll
    for (int nt = 0; nt < 19; ++nt) {
      int v = nt * 16 + n;
      float e = (v < U_) ? __expf(l[nt][r] - mx) : 0.f;
      l[nt][r] = e;
      s += e;
    }
    for (int off = 1; off < 16; off <<= 1) s += __shfl_xor(s, off, 32);
    float inv = rowscale / s;
    if (u < U_) {
      #pragma unroll
      for (int nt = 0; nt < 19; ++nt) {
        int v = nt * 16 + n;
        if (v < U_) atomicAdd(attention + (size_t)u * U_ + v, l[nt][r] * inv);
      }
    }
  }
}

// ---------------------------------------------------------------------------
// K5: Xn = relu(wres @ fc_ta_w^T + b) into Xn_full[:, 0:128] (row stride 320)
// ---------------------------------------------------------------------------
__global__ __launch_bounds__(256) void xn_kernel(
    const bf16* __restrict__ wres, const bf16* __restrict__ fctaw_bf,
    const float* __restrict__ fcta_b, bf16* __restrict__ xnf) {
  int wave = (blockIdx.x * 256 + threadIdx.x) >> 5;  // 600 waves
  int lane = threadIdx.x & 31;
  int i0 = wave * 16;
  v16bf af[4];
  #pragma unroll
  for (int ks = 0; ks < 4; ++ks)
    af[ks] = load_a_frag(wres + (size_t)i0 * 128 + ks * 32, 128, lane);
  int n = lane & 15, half = lane >> 4;
  #pragma unroll
  for (int nt = 0; nt < 8; ++nt) {
    v8f acc = v8f_zero();
    #pragma unroll
    for (int ks = 0; ks < 4; ++ks)
      acc = wmma_bf(af[ks], load_b_frag(fctaw_bf + (nt * 16) * 128 + ks * 32, 128, lane), acc);
    int c = nt * 16 + n;
    float bv = fcta_b[c];
    #pragma unroll
    for (int r = 0; r < 8; ++r)
      xnf[(size_t)(i0 + half * 8 + r) * 320 + c] = (bf16)fmaxf(acc[r] + bv, 0.f);
  }
}

// K5b: emb[u][e] = sem_embs[u] . sem_w[e] + sem_b[e]
__global__ void emb_kernel(const float* __restrict__ se, const float* __restrict__ sw,
                           const float* __restrict__ sb, float* __restrict__ emb) {
  int i = blockIdx.x * 256 + threadIdx.x;
  if (i >= U_ * E_) return;
  int u = i / E_, e = i % E_;
  float acc = sb[e];
  for (int k = 0; k < E_; ++k) acc += se[u * E_ + k] * sw[e * E_ + k];
  emb[i] = acc;
}

// K5c: broadcast emb into Xn_full cols 128..295 (bf16), zero pad 296..319
__global__ void fill_kernel(const float* __restrict__ emb, bf16* __restrict__ xnf) {
  int i = blockIdx.x * 256 + threadIdx.x;
  if (i >= BU * 192) return;
  int row = i / 192, cc = i % 192;
  int u = row % U_;
  float v = (cc < E_) ? emb[u * E_ + cc] : 0.f;
  xnf[(size_t)row * 320 + 128 + cc] = (bf16)v;
}

// ---------------------------------------------------------------------------
// K6: adjacency pipeline (small, VALU)
// ---------------------------------------------------------------------------
__global__ void mag_kernel(const float* __restrict__ emb, float* __restrict__ mag) {
  int u = blockIdx.x * 256 + threadIdx.x;
  if (u >= U_) return;
  float s = 0.f;
  for (int e = 0; e < E_; ++e) { float v = emb[u * E_ + e]; s += v * v; }
  mag[u] = sqrtf(s);
}

__global__ __launch_bounds__(512) void attrow_kernel(
    const float* __restrict__ emb, const float* __restrict__ mag,
    const float* __restrict__ attention, const float* __restrict__ dist,
    const float* __restrict__ alpha_p, float* __restrict__ att,
    float* __restrict__ rowmax) {
  int i = blockIdx.x;
  int j = threadIdx.x;
  __shared__ float smax[512];
  float a = -1e30f;
  if (j < U_) {
    float alpha = alpha_p[0];
    float dot = 0.f;
    for (int e = 0; e < E_; ++e) dot += emb[i * E_ + e] * emb[j * E_ + e];
    float ea = fabsf(dot / (mag[i] * mag[j]));
    float v = (alpha * dist[i * U_ + j] + (1.f - alpha) * ea) * attention[i * U_ + j];
    att[i * U_ + j] = v;
    a = v;
  }
  smax[j] = a;
  __syncthreads();
  for (int s = 256; s > 0; s >>= 1) {
    if (j < s) smax[j] = fmaxf(smax[j], smax[j + s]);
    __syncthreads();
  }
  if (j == 0) rowmax[i] = smax[0];
}

// threshold + emit att_thr (second output)
__global__ void thr_kernel(const float* __restrict__ att, const float* __restrict__ rowmax,
                           float* __restrict__ att_thr, float* __restrict__ out_attthr) {
  int i = blockIdx.x * 256 + threadIdx.x;
  if (i >= U_ * U_) return;
  int r = i / U_;
  float a = att[i];
  float ratio = a / rowmax[r];
  float v = (powf(ratio, 2.5f) > 0.15f) ? a : 0.f;
  att_thr[i] = v;
  out_attthr[i] = v;
}

__global__ void deg_kernel(const float* __restrict__ att_thr, float* __restrict__ dinv) {
  int j = blockIdx.x * 256 + threadIdx.x;
  if (j >= U_) return;
  float s = 1.f;                               // + identity self loop
  for (int i = 0; i < U_; ++i) s += att_thr[i * U_ + j];
  dinv[j] = (s > 0.f) ? rsqrtf(s) : 0.f;
}

__global__ void norm_kernel(const float* __restrict__ att_thr, const float* __restrict__ dinv,
                            float* __restrict__ norm) {
  int idx = blockIdx.x * 256 + threadIdx.x;
  if (idx >= U_ * U_) return;
  int i = idx / U_, j = idx % U_;
  float a = att_thr[idx] + ((i == j) ? 1.f : 0.f);
  norm[idx] = dinv[i] * a * dinv[j];
}

// ---------------------------------------------------------------------------
// K7: GCN. Feature GEMM (N=32) on WMMA; adjacency matmul (1.5% FLOPs) VALU.
// ---------------------------------------------------------------------------
template <int KSTEPS>
__global__ __launch_bounds__(256) void gemm_n32_kernel(
    const bf16* __restrict__ A, int lda, const bf16* __restrict__ W, int ldw,
    bf16* __restrict__ out) {
  int wave = (blockIdx.x * 256 + threadIdx.x) >> 5;  // 600 waves
  int lane = threadIdx.x & 31;
  int i0 = wave * 16;
  v16bf af[KSTEPS];
  #pragma unroll
  for (int ks = 0; ks < KSTEPS; ++ks)
    af[ks] = load_a_frag(A + (size_t)i0 * lda + ks * 32, lda, lane);
  int n = lane & 15, half = lane >> 4;
  #pragma unroll
  for (int nt = 0; nt < 2; ++nt) {
    v8f acc = v8f_zero();
    #pragma unroll
    for (int ks = 0; ks < KSTEPS; ++ks)
      acc = wmma_bf(af[ks], load_b_frag(W + (nt * 16) * ldw + ks * 32, ldw, lane), acc);
    int c = nt * 16 + n;
    #pragma unroll
    for (int r = 0; r < 8; ++r)
      out[(size_t)(i0 + half * 8 + r) * 32 + c] = (bf16)acc[r];
  }
}

__global__ void gcn_adj_kernel(const float* __restrict__ norm, const bf16* __restrict__ tmp,
                               const float* __restrict__ bias, bf16* __restrict__ g) {
  int idx = blockIdx.x * 256 + threadIdx.x;     // B*U*32 = 307200
  if (idx >= B_ * U_ * CG_) return;
  int f = idx & 31;
  int j = (idx >> 5) % U_;
  int b = idx / (U_ * CG_);
  float acc = bias[f];
  for (int i = 0; i < U_; ++i)
    acc += norm[i * U_ + j] * (float)tmp[((size_t)b * U_ + i) * 32 + f];
  g[((size_t)b * U_ + j) * 32 + f] = (bf16)fmaxf(acc, 0.f);
}

// ---------------------------------------------------------------------------
// K8: readout. fc1 (K=352 padded: 10 ksteps from Xn_full + 1 from g) with
// relu, fused fc2 row-dot (N=1) via cross-lane reduce. Writes forecast.
// ---------------------------------------------------------------------------
__global__ __launch_bounds__(128) void fc_kernel(
    const bf16* __restrict__ xnf, const bf16* __restrict__ g,
    const bf16* __restrict__ fc1w_bf, const float* __restrict__ fc1_b,
    const float* __restrict__ fc2_w, const float* __restrict__ fc2_b,
    float* __restrict__ out) {
  int wave = (blockIdx.x * 128 + threadIdx.x) >> 5;  // 600 waves
  int lane = threadIdx.x & 31;
  int i0 = wave * 16;
  v16bf af[11];
  #pragma unroll
  for (int ks = 0; ks < 10; ++ks)
    af[ks] = load_a_frag(xnf + (size_t)i0 * 320 + ks * 32, 320, lane);
  af[10] = load_a_frag(g + (size_t)i0 * 32, 32, lane);
  int n = lane & 15, half = lane >> 4;
  float rs[8];
  #pragma unroll
  for (int r = 0; r < 8; ++r) rs[r] = 0.f;
  for (int nt = 0; nt < 19; ++nt) {
    v8f acc = v8f_zero();
    #pragma unroll
    for (int ks = 0; ks < 11; ++ks)
      acc = wmma_bf(af[ks], load_b_frag(fc1w_bf + (nt * 16) * 352 + ks * 32, 352, lane), acc);
    int c = nt * 16 + n;
    float bv = (c < 296) ? fc1_b[c] : 0.f;
    float w2 = (c < 296) ? fc2_w[c] : 0.f;
    #pragma unroll
    for (int r = 0; r < 8; ++r) rs[r] += fmaxf(acc[r] + bv, 0.f) * w2;
  }
  #pragma unroll
  for (int r = 0; r < 8; ++r) {
    float v = rs[r];
    for (int off = 1; off < 16; off <<= 1) v += __shfl_xor(v, off, 32);
    if (n == 0) out[i0 + half * 8 + r] = v + fc2_b[0];
  }
}

// ---------------------------------------------------------------------------
// Host: workspace carve-out + launch graph (all on `stream`)
// ---------------------------------------------------------------------------
extern "C" void kernel_launch(void* const* d_in, const int* in_sizes, int n_in,
                              void* d_out, int out_size, void* d_ws, size_t ws_size,
                              hipStream_t stream) {
  (void)in_sizes; (void)n_in; (void)out_size; (void)ws_size;
  const float* x        = (const float*)d_in[0];
  const float* s1w1     = (const float*)d_in[1];
  const float* s1b1     = (const float*)d_in[2];
  const float* s1w2     = (const float*)d_in[3];
  const float* s1b2     = (const float*)d_in[4];
  const float* gwih     = (const float*)d_in[5];
  const float* gbih     = (const float*)d_in[6];
  const float* gwhh     = (const float*)d_in[7];
  const float* gbhh     = (const float*)d_in[8];
  const float* attn_w   = (const float*)d_in[9];
  const float* attn_b   = (const float*)d_in[10];
  const float* attn_v   = (const float*)d_in[11];
  const float* ln_w     = (const float*)d_in[12];
  const float* ln_b     = (const float*)d_in[13];
  const float* mha_w    = (const float*)d_in[14];
  const float* mha_b    = (const float*)d_in[15];
  const float* fcta_w   = (const float*)d_in[16];
  const float* fcta_b   = (const float*)d_in[17];
  const float* sem_embs = (const float*)d_in[18];
  const float* sem_w    = (const float*)d_in[19];
  const float* sem_b    = (const float*)d_in[20];
  const float* alpha_p  = (const float*)d_in[21];
  const float* dist     = (const float*)d_in[22];
  const float* gw0      = (const float*)d_in[23];
  const float* gb0      = (const float*)d_in[24];
  const float* gw1      = (const float*)d_in[25];
  const float* gb1      = (const float*)d_in[26];
  const float* gw2      = (const float*)d_in[27];
  const float* gb2      = (const float*)d_in[28];
  const float* fc1w     = (const float*)d_in[29];
  const float* fc1b     = (const float*)d_in[30];
  const float* fc2w     = (const float*)d_in[31];
  const float* fc2b     = (const float*)d_in[32];
  float* out = (float*)d_out;   // [0,9600): forecast; [9600,99600): att_thr

  char* base = (char*)d_ws;
  size_t off = 0;
  auto alloc = [&](size_t bytes) -> void* {
    off = (off + 255) & ~(size_t)255;
    void* p = base + off;
    off += bytes;
    return p;
  };
  bf16* feats    = (bf16*)alloc((size_t)MT_ * 128 * 2);
  bf16* outs     = (bf16*)alloc((size_t)MT_ * 128 * 2);
  bf16* wih_bf   = (bf16*)alloc((size_t)U_ * 384 * 128 * 2);
  bf16* whh_bf   = (bf16*)alloc((size_t)U_ * 384 * 128 * 2);
  bf16* s1w2_bf  = (bf16*)alloc(128 * 64 * 2);
  bf16* attnw_bf = (bf16*)alloc(128 * 256 * 2);
  bf16* mhaw_bf  = (bf16*)alloc(384 * 128 * 2);
  bf16* fctaw_bf = (bf16*)alloc(128 * 128 * 2);
  bf16* gw0_bf   = (bf16*)alloc(32 * 320 * 2);
  bf16* gw1_bf   = (bf16*)alloc(32 * 32 * 2);
  bf16* gw2_bf   = (bf16*)alloc(32 * 32 * 2);
  bf16* fc1w_bf  = (bf16*)alloc(304 * 352 * 2);
  float* scores  = (float*)alloc((size_t)MT_ * 4);
  bf16* wres     = (bf16*)alloc((size_t)BU * 128 * 2);
  bf16* qb       = (bf16*)alloc((size_t)BU * 128 * 2);
  bf16* kb       = (bf16*)alloc((size_t)BU * 128 * 2);
  float* attention = (float*)alloc((size_t)U_ * U_ * 4);
  float* emb     = (float*)alloc((size_t)U_ * E_ * 4);
  float* mag     = (float*)alloc(U_ * 4);
  bf16* xnf      = (bf16*)alloc((size_t)BU * 320 * 2);
  float* att     = (float*)alloc((size_t)U_ * U_ * 4);
  float* rowmax  = (float*)alloc(U_ * 4);
  float* att_thr = (float*)alloc((size_t)U_ * U_ * 4);
  float* dinv    = (float*)alloc(U_ * 4);
  float* normadj = (float*)alloc((size_t)U_ * U_ * 4);
  bf16* tmpb     = (bf16*)alloc((size_t)BU * 32 * 2);
  bf16* ga       = (bf16*)alloc((size_t)BU * 32 * 2);
  bf16* gb       = (bf16*)alloc((size_t)BU * 32 * 2);

  auto grid1 = [](int n) { return (n + 255) / 256; };

  // --- weight conversions to bf16 ---
  int nwih = U_ * 384 * 128;
  cvt_kernel<<<grid1(nwih), 256, 0, stream>>>(gwih, wih_bf, nwih);
  cvt_kernel<<<grid1(nwih), 256, 0, stream>>>(gwhh, whh_bf, nwih);
  cvt_kernel<<<grid1(128 * 64), 256, 0, stream>>>(s1w2, s1w2_bf, 128 * 64);
  cvt_kernel<<<grid1(128 * 256), 256, 0, stream>>>(attn_w, attnw_bf, 128 * 256);
  cvt_kernel<<<grid1(384 * 128), 256, 0, stream>>>(mha_w, mhaw_bf, 384 * 128);
  cvt_kernel<<<grid1(128 * 128), 256, 0, stream>>>(fcta_w, fctaw_bf, 128 * 128);
  cvt_pad_kernel<<<grid1(32 * 320), 256, 0, stream>>>(gw0, gw0_bf, 32, 296, 320);
  cvt_kernel<<<grid1(32 * 32), 256, 0, stream>>>(gw1, gw1_bf, 32 * 32);
  cvt_kernel<<<grid1(32 * 32), 256, 0, stream>>>(gw2, gw2_bf, 32 * 32);
  cvt_fc1_kernel<<<grid1(304 * 352), 256, 0, stream>>>(fc1w, fc1w_bf);

  // --- pipeline ---
  seq1_kernel<<<1800, 256, 0, stream>>>(x, s1w1, s1b1, s1w2_bf, s1b2, feats);
  gru_kernel<<<U_, 256, 0, stream>>>(feats, wih_bf, whh_bf, gbih, gbhh, outs);
  score_kernel<<<1800, 256, 0, stream>>>(outs, attnw_bf, attn_b, attn_v, scores);
  timeattn_kernel<<<BU, 128, 0, stream>>>(scores, outs, ln_w, ln_b, wres);
  qk_kernel<<<150, 256, 0, stream>>>(wres, mhaw_bf, mha_b, qb, kb);
  zero_f32_kernel<<<grid1(U_ * U_), 256, 0, stream>>>(attention, U_ * U_);
  mha_attn_kernel<<<1216, 128, 0, stream>>>(qb, kb, attention);
  xn_kernel<<<75, 256, 0, stream>>>(wres, fctaw_bf, fcta_b, xnf);
  emb_kernel<<<grid1(U_ * E_), 256, 0, stream>>>(sem_embs, sem_w, sem_b, emb);
  fill_kernel<<<grid1(BU * 192), 256, 0, stream>>>(emb, xnf);
  mag_kernel<<<grid1(U_), 256, 0, stream>>>(emb, mag);
  attrow_kernel<<<U_, 512, 0, stream>>>(emb, mag, attention, dist, alpha_p, att, rowmax);
  thr_kernel<<<grid1(U_ * U_), 256, 0, stream>>>(att, rowmax, att_thr, out + BU);
  deg_kernel<<<grid1(U_), 256, 0, stream>>>(att_thr, dinv);
  norm_kernel<<<grid1(U_ * U_), 256, 0, stream>>>(att_thr, dinv, normadj);

  // GCN: 3 layers (WMMA feature GEMM + VALU adjacency apply)
  gemm_n32_kernel<10><<<75, 256, 0, stream>>>(xnf, 320, gw0_bf, 320, tmpb);
  gcn_adj_kernel<<<grid1(B_ * U_ * CG_), 256, 0, stream>>>(normadj, tmpb, gb0, ga);
  gemm_n32_kernel<1><<<75, 256, 0, stream>>>(ga, 32, gw1_bf, 32, tmpb);
  gcn_adj_kernel<<<grid1(B_ * U_ * CG_), 256, 0, stream>>>(normadj, tmpb, gb1, gb);
  gemm_n32_kernel<1><<<75, 256, 0, stream>>>(gb, 32, gw2_bf, 32, tmpb);
  gcn_adj_kernel<<<grid1(B_ * U_ * CG_), 256, 0, stream>>>(normadj, tmpb, gb2, ga);

  // readout
  fc_kernel<<<150, 128, 0, stream>>>(xnf, ga, fc1w_bf, fc1b, fc2w, fc2b, out);
}
